// SegmentFusion_83992380441201
// MI455X (gfx1250) — compile-verified, matrix operand order checked by reading, not run on hardware
//
#include <hip/hip_runtime.h>

typedef __attribute__((ext_vector_type(16))) __bf16 v16bf;
typedef __attribute__((ext_vector_type(8)))  __bf16 v8bf;
typedef __attribute__((ext_vector_type(8)))  float  v8f;

__device__ __forceinline__ unsigned short f32_to_bf16(float f) {
  unsigned int u = __float_as_uint(f);
  u += 0x7FFFu + ((u >> 16) & 1u);           // round-to-nearest-even
  return (unsigned short)(u >> 16);
}

// Repack W (512x256 f32, row-major) into Bcol[n][k] bf16, k contiguous.
// n < 256  -> W_top column n  (weights for feat_a)
// n >= 256 -> W_bot column n-256 (weights for feat_b)
__global__ __launch_bounds__(256) void prep_weights(const float* __restrict__ W,
                                                    unsigned short* __restrict__ Bcol) {
  int idx = blockIdx.x * 256 + threadIdx.x;  // 0..131071
  int n = idx >> 8;
  int k = idx & 255;
  float v = (n < 256) ? W[k * 256 + n] : W[(256 + k) * 256 + (n - 256)];
  Bcol[n * 256 + k] = f32_to_bf16(v);
}

__global__ __launch_bounds__(256) void fused_gemm_ln_relu(
    const float* __restrict__ nf,            // (B,3,256) f32
    const unsigned short* __restrict__ Bcol, // (512,256) bf16 col-major-of-W
    const float* __restrict__ bias,          // (256)
    const float* __restrict__ gamma,         // (256)
    const float* __restrict__ beta,          // (256)
    float* __restrict__ out) {               // (B,5,256) f32
  __shared__ unsigned short sA[48 * 256];    // 24 KB  A tile (bf16)
  __shared__ float          sP[48 * 512];    // 96 KB  GEMM out: cols 0..255 = Y, 256..511 = Z

  const int tid  = threadIdx.x;
  const int wid  = tid >> 5;
  const int lane = tid & 31;
  const int half = lane >> 4;
  const int l16  = lane & 15;

  // ---- Stage A tile: 48 contiguous rows (16 batches x 3 nodes) fp32 -> bf16 ----
  const float4* src = (const float4*)(nf + (size_t)blockIdx.x * 48 * 256);
  #pragma unroll
  for (int i = tid; i < (48 * 256) / 4; i += 256) {
    float4 v = src[i];
    unsigned int p0 = (unsigned int)f32_to_bf16(v.x) | ((unsigned int)f32_to_bf16(v.y) << 16);
    unsigned int p1 = (unsigned int)f32_to_bf16(v.z) | ((unsigned int)f32_to_bf16(v.w) << 16);
    *(uint2*)(&sA[i * 4]) = make_uint2(p0, p1);
  }
  __syncthreads();

  // ---- GEMM: 96 tiles of 16x16 (mt 0..2, nt 0..31).
  // Each wave owns 4 nt columns; B fragments for a column are loaded from
  // global once into registers and reused across the 3 mt tiles (3x less
  // L0/L2 traffic than tile-major order).
  for (int nt4 = 0; nt4 < 4; ++nt4) {
    const int nt = wid * 4 + nt4;
    // Lane's bf16 fragment addressing per ISA 16-bit layouts:
    // elems 0..7 -> K = kc*32 + half*8 .. +7 ; elems 8..15 -> +16
    const unsigned short* bB = &Bcol[(nt * 16 + l16) * 256 + half * 8];
    v16bf bf[8];
    #pragma unroll
    for (int kc = 0; kc < 8; ++kc) {
      v8bf blo = *(const v8bf*)(bB + kc * 32);
      v8bf bhi = *(const v8bf*)(bB + kc * 32 + 16);
      bf[kc] = __builtin_shufflevector(blo, bhi, 0,1,2,3,4,5,6,7,8,9,10,11,12,13,14,15);
    }
    for (int mt = 0; mt < 3; ++mt) {
      const unsigned short* aB = &sA[(mt * 16 + l16) * 256 + half * 8];
      v8f c = {};
      #pragma unroll
      for (int kc = 0; kc < 8; ++kc) {
        v8bf alo = *(const v8bf*)(aB + kc * 32);
        v8bf ahi = *(const v8bf*)(aB + kc * 32 + 16);
        v16bf a = __builtin_shufflevector(alo, ahi, 0,1,2,3,4,5,6,7,8,9,10,11,12,13,14,15);
        c = __builtin_amdgcn_wmma_f32_16x16x32_bf16(false, a, false, bf[kc], (short)0, c,
                                                    false, false);
      }
      // C/D layout: VGPR v, lane l -> M = v + 8*(l>=16), N = l&15
      #pragma unroll
      for (int v = 0; v < 8; ++v)
        sP[(mt * 16 + half * 8 + v) * 512 + nt * 16 + l16] = c[v];
    }
  }
  __syncthreads();

  // ---- Epilogue: 80 output rows (16 batches x 5 pipes), 10 per wave ----
  const float4* bias4 = (const float4*)bias;
  const float4* g4    = (const float4*)gamma;
  const float4* be4   = (const float4*)beta;
  const int f0 = lane * 8;

  for (int i = 0; i < 10; ++i) {
    const int o  = wid * 10 + i;          // 0..79
    const int bl = o / 5;
    const int p  = o - bl * 5;
    const int ra = bl * 3 + ((0x22100 >> (4 * p)) & 0xF);  // IDX_A = {0,0,1,2,2}
    const int rb = bl * 3 + ((0x10210 >> (4 * p)) & 0xF);  // IDX_B = {0,1,2,0,1}

    const float4 y0 = *(const float4*)&sP[ra * 512 + f0];
    const float4 y1 = *(const float4*)&sP[ra * 512 + f0 + 4];
    const float4 z0 = *(const float4*)&sP[rb * 512 + 256 + f0];
    const float4 z1 = *(const float4*)&sP[rb * 512 + 256 + f0 + 4];
    const float4 b0 = bias4[lane * 2];
    const float4 b1 = bias4[lane * 2 + 1];

    float vv[8];
    vv[0] = y0.x + z0.x + b0.x;  vv[1] = y0.y + z0.y + b0.y;
    vv[2] = y0.z + z0.z + b0.z;  vv[3] = y0.w + z0.w + b0.w;
    vv[4] = y1.x + z1.x + b1.x;  vv[5] = y1.y + z1.y + b1.y;
    vv[6] = y1.z + z1.z + b1.z;  vv[7] = y1.w + z1.w + b1.w;

    float s1 = 0.f, s2 = 0.f;
    #pragma unroll
    for (int j = 0; j < 8; ++j) { s1 += vv[j]; s2 += vv[j] * vv[j]; }
    #pragma unroll
    for (int off = 16; off >= 1; off >>= 1) {
      s1 += __shfl_xor(s1, off, 32);
      s2 += __shfl_xor(s2, off, 32);
    }
    const float mean = s1 * (1.0f / 256.0f);
    const float var  = s2 * (1.0f / 256.0f) - mean * mean;
    const float rstd = rsqrtf(var + 1e-5f);

    const float4 g0 = g4[lane * 2],  g1 = g4[lane * 2 + 1];
    const float4 t0 = be4[lane * 2], t1 = be4[lane * 2 + 1];
    float ga[8] = {g0.x, g0.y, g0.z, g0.w, g1.x, g1.y, g1.z, g1.w};
    float bt[8] = {t0.x, t0.y, t0.z, t0.w, t1.x, t1.y, t1.z, t1.w};

    float r[8];
    #pragma unroll
    for (int j = 0; j < 8; ++j)
      r[j] = fmaxf(0.0f, (vv[j] - mean) * rstd * ga[j] + bt[j]);

    float4* dst = (float4*)(out + (((size_t)blockIdx.x * 16 + bl) * 5 + p) * 256 + f0);
    dst[0] = make_float4(r[0], r[1], r[2], r[3]);
    dst[1] = make_float4(r[4], r[5], r[6], r[7]);
  }
}

extern "C" void kernel_launch(void* const* d_in, const int* in_sizes, int n_in,
                              void* d_out, int out_size, void* d_ws, size_t ws_size,
                              hipStream_t stream) {
  const float* nf    = (const float*)d_in[0];
  const float* W     = (const float*)d_in[1];
  const float* bias  = (const float*)d_in[2];
  const float* gamma = (const float*)d_in[3];
  const float* beta  = (const float*)d_in[4];
  float* out = (float*)d_out;
  unsigned short* wsB = (unsigned short*)d_ws;  // needs 512*256*2 = 256 KB

  // Repack/convert W every call (deterministic, graph-capture safe).
  prep_weights<<<512, 256, 0, stream>>>(W, wsB);

  const int n_batches = in_sizes[0] / (3 * 256);   // 131072
  fused_gemm_ln_relu<<<n_batches / 16, 256, 0, stream>>>(nf, wsB, bias, gamma, beta, out);
}
